// _DSConv_53584011985475
// MI455X (gfx1250) — 8-bit matrix operand order repaired and checked by reading, compiled, not run on hardware
//
#include <hip/hip_runtime.h>
#include <math.h>

typedef int v8i __attribute__((ext_vector_type(8)));

#define W_QMAX 7.0f
#define A_QMAX 15.0f
#define BN_EPS 1e-5f

#define NB   32
#define CI   256
#define CO   512
#define HH   64
#define WW   64
#define NPOS (NB * HH * WW)
#define PTILES (NPOS / 16)
#define KTILES (CI / 64)
#define MTILES (CO / 16)
#define NGROUPS (PTILES / 4)
#define MGROUPS (MTILES / 2)

#define WS_BN1A   0u
#define WS_BN1B   1024u
#define WS_DWQ    2048u
#define WS_A2     12288u
#define WS_B2     14336u
#define WS_WFRAG  16384u
#define WS_XFRAG  147456u

__global__ void k_prep_dw(const float* __restrict__ w_dw,
                          const float* __restrict__ g1, const float* __restrict__ b1,
                          const float* __restrict__ m1, const float* __restrict__ v1,
                          char* __restrict__ ws) {
  int c = threadIdx.x;
  float* dwq = (float*)(ws + WS_DWQ);
  float* bnA = (float*)(ws + WS_BN1A);
  float* bnB = (float*)(ws + WS_BN1B);
  float w[9], mx = 0.f;
#pragma unroll
  for (int i = 0; i < 9; i++) { w[i] = w_dw[c * 9 + i]; mx = fmaxf(mx, fabsf(w[i])); }
  float scale = fmaxf(mx / W_QMAX, 1e-8f);
#pragma unroll
  for (int i = 0; i < 9; i++) {
    float q = fminf(fmaxf(rintf(w[i] / scale), -W_QMAX), W_QMAX);
    dwq[c * 9 + i] = q * scale;
  }
  float inv = g1[c] * rsqrtf(v1[c] + BN_EPS);
  bnA[c] = inv;
  bnB[c] = b1[c] - m1[c] * inv;
}

__global__ void k_prep_pw(const float* __restrict__ w_pw,
                          const float* __restrict__ g2, const float* __restrict__ b2,
                          const float* __restrict__ m2, const float* __restrict__ v2,
                          const float* __restrict__ s1p,
                          char* __restrict__ ws) {
  __shared__ float red[256];
  int o = blockIdx.x;
  int c = threadIdx.x;
  float w = w_pw[o * CI + c];
  red[c] = fabsf(w);
  __syncthreads();
  for (int s = 128; s > 0; s >>= 1) {
    if (c < s) red[c] = fmaxf(red[c], red[c + s]);
    __syncthreads();
  }
  float scale = fmaxf(red[0] / W_QMAX, 1e-8f);
  int qi = (int)fminf(fmaxf(rintf(w / scale), -W_QMAX), W_QMAX);

  int tm = o >> 4, oc16 = o & 15;
  int tk = c >> 6, k = c & 63;
  int lane = oc16 + 16 * ((k >> 3) & 1);
  int v = ((k >> 2) & 1) | (((k >> 4) & 1) << 1) | (((k >> 5) & 1) << 2);
  int j = v * 4 + (k & 3);
  signed char* wf = (signed char*)(ws + WS_WFRAG);
  wf[(size_t)(tm * KTILES + tk) * 1024 + lane * 32 + j] = (signed char)qi;

  if (c == 0) {
    float inv = g2[o] * rsqrtf(v2[o] + BN_EPS);
    ((float*)(ws + WS_A2))[o] = scale * s1p[0] * inv;
    ((float*)(ws + WS_B2))[o] = b2[o] - m2[o] * inv;
  }
}

__global__ void k_dwconv(const float* __restrict__ x,
                         const float* __restrict__ s1p,
                         char* __restrict__ ws) {
  int tid = threadIdx.x;
  int xc = tid & 63;
  int cl = tid >> 6;
  int n = blockIdx.x >> 6;
  int y = blockIdx.x & 63;
  int c = blockIdx.y * 4 + cl;

  const float* dwq = (const float*)(ws + WS_DWQ) + c * 9;
  float bnA = ((const float*)(ws + WS_BN1A))[c];
  float bnB = ((const float*)(ws + WS_BN1B))[c];
  const float* xp = x + ((size_t)n * CI + c) * (HH * WW);

  float acc = 0.f;
#pragma unroll
  for (int ky = 0; ky < 3; ky++) {
    int yy = y + ky - 1;
    if (yy < 0 || yy >= HH) continue;
#pragma unroll
    for (int kx = 0; kx < 3; kx++) {
      int xx = xc + kx - 1;
      if (xx < 0 || xx >= WW) continue;
      acc = fmaf(xp[yy * WW + xx], dwq[ky * 3 + kx], acc);
    }
  }
  float t = fmaxf(fmaf(acc, bnA, bnB), 0.f);
  int q = (int)fminf(rintf(t / s1p[0]), A_QMAX);

  int pos = n * 4096 + y * 64 + xc;
  int tp = pos >> 4, p16 = pos & 15;
  int tk = c >> 6, k = c & 63;
  int lane = p16 + 16 * ((k >> 3) & 1);
  int j = ((k >> 5) << 4) + (((k >> 4) & 1) << 3) + (k & 7);
  unsigned char* xf = (unsigned char*)(ws + WS_XFRAG);
  xf[(size_t)(tp * KTILES + tk) * 1024 + lane * 32 + j] = (unsigned char)q;
}

__global__ void __launch_bounds__(256)
k_pwgemm(const char* __restrict__ ws, const float* __restrict__ s2p,
         float* __restrict__ out) {
  int lane = threadIdx.x & 31;
  int wid = blockIdx.x * 8 + (threadIdx.x >> 5);
  int ngroup = wid & (NGROUPS - 1);
  int mgroup = wid >> 11;

  const v8i* wf = (const v8i*)(ws + WS_WFRAG);
  const v8i* xf = (const v8i*)(ws + WS_XFRAG);
  const float* A2 = (const float*)(ws + WS_A2);
  const float* B2 = (const float*)(ws + WS_B2);
  float s2 = s2p[0];
  float inv_s2 = 1.0f / s2;

  int mt0 = mgroup * 2;
  int nt0 = ngroup * 4;

  v8i acc[2][4];
  const v8i vzero = {0, 0, 0, 0, 0, 0, 0, 0};
#pragma unroll
  for (int m = 0; m < 2; m++)
#pragma unroll
    for (int n = 0; n < 4; n++) acc[m][n] = vzero;

#pragma unroll
  for (int k = 0; k < KTILES; k++) {
    v8i wv[2], xv[4];
#pragma unroll
    for (int m = 0; m < 2; m++)
      wv[m] = wf[((size_t)(mt0 + m) * KTILES + k) * 32 + lane];
#pragma unroll
    for (int n = 0; n < 4; n++)
      xv[n] = xf[((size_t)(nt0 + n) * KTILES + k) * 32 + lane];
    if (k + 1 < KTILES)
      __builtin_prefetch((const char*)&xf[((size_t)nt0 * KTILES + k + 1) * 32 + lane], 0, 0);
#pragma unroll
    for (int m = 0; m < 2; m++)
#pragma unroll
      for (int n = 0; n < 4; n++)
        acc[m][n] = __builtin_amdgcn_wmma_i32_16x16x64_iu8(
             true, wv[m],
             false, xv[n],
            acc[m][n],  false,  false);
  }

  int lhi = lane >> 4;
  int l15 = lane & 15;
#pragma unroll
  for (int m = 0; m < 2; m++) {
    int ocb = (mt0 + m) * 16 + 8 * lhi;
    float a2[8], b2c[8];
#pragma unroll
    for (int r = 0; r < 8; r++) { a2[r] = A2[ocb + r]; b2c[r] = B2[ocb + r]; }
#pragma unroll
    for (int n = 0; n < 4; n++) {
      int pos = (nt0 + n) * 16 + l15;
      int nb = pos >> 12;
      int sp = pos & 4095;
      float* op = out + (size_t)nb * CO * 4096 + sp;
#pragma unroll
      for (int r = 0; r < 8; r++) {
        float y = fmaf((float)acc[m][n][r], a2[r], b2c[r]);
        y = fmaxf(y, 0.f);
        float q = fminf(rintf(y * inv_s2), A_QMAX);
        op[(size_t)(ocb + r) * 4096] = q * s2;
      }
    }
  }
}

extern "C" void kernel_launch(void* const* d_in, const int* in_sizes, int n_in,
                              void* d_out, int out_size, void* d_ws, size_t ws_size,
                              hipStream_t stream) {
  const float* x    = (const float*)d_in[0];
  const float* w_dw = (const float*)d_in[1];
  const float* w_pw = (const float*)d_in[2];
  const float* g1   = (const float*)d_in[3];
  const float* b1   = (const float*)d_in[4];
  const float* g2   = (const float*)d_in[5];
  const float* b2   = (const float*)d_in[6];
  const float* s1   = (const float*)d_in[7];
  const float* s2   = (const float*)d_in[8];
  const float* m1   = (const float*)d_in[9];
  const float* v1   = (const float*)d_in[10];
  const float* m2   = (const float*)d_in[11];
  const float* v2   = (const float*)d_in[12];
  char* ws = (char*)d_ws;
  float* out = (float*)d_out;

  k_prep_dw<<<1, 256, 0, stream>>>(w_dw, g1, b1, m1, v1, ws);
  k_prep_pw<<<CO, 256, 0, stream>>>(w_pw, g2, b2, m2, v2, s1, ws);
  dim3 gdw(NB * HH, CI / 4);
  k_dwconv<<<gdw, 256, 0, stream>>>(x, s1, ws);
  int gemm_blocks = (MGROUPS * NGROUPS) / 8;
  k_pwgemm<<<gemm_blocks, 256, 0, stream>>>(ws, s2, out);
}
